// Decoder_87127706567207
// MI455X (gfx1250) — matrix operand order repaired and checked by reading, not compiled, not run on hardware
//
#include <hip/hip_runtime.h>
#include <hip/hip_bf16.h>

// ---------------------------------------------------------------------------
// Decoder (TCN + degenerate attention + LSTM scan) for MI455X / gfx1250.
// B=1024, T=64 (Tm1=63), F=15, H=512.  All GEMMs via v_wmma_f32_16x16x32_bf16.
// Register-blocked 4x along M (conv) / batch (LSTM) to cut L2 traffic 4x.
// ---------------------------------------------------------------------------

typedef __attribute__((ext_vector_type(16))) __bf16 v16bf;
typedef __attribute__((ext_vector_type(8)))  __bf16 v8bf;
typedef __attribute__((ext_vector_type(8)))  float  v8f;

#define SHUF16(lo, hi) __builtin_shufflevector(lo, hi, 0,1,2,3,4,5,6,7,8,9,10,11,12,13,14,15)
#define WMMA_BF16(av, bv, cc) \
  __builtin_amdgcn_wmma_f32_16x16x32_bf16(false, (av), false, (bv), (short)0, (cc), false, false)

__device__ __forceinline__ float sigmoidf_(float x) { return 1.0f / (1.0f + __expf(-x)); }

// ---------------- elementwise helpers ----------------
__global__ void k_f32_to_bf16(const float* __restrict__ in, __bf16* __restrict__ out, int n) {
  int i = blockIdx.x * blockDim.x + threadIdx.x;
  if (i < n) out[i] = (__bf16)in[i];
}

__global__ void k_zero_u32(unsigned int* __restrict__ p, int n) {
  int i = blockIdx.x * blockDim.x + threadIdx.x;
  if (i < n) p[i] = 0u;
}

__global__ void k_add_relu_bf(const __bf16* __restrict__ h, const __bf16* __restrict__ r,
                              __bf16* __restrict__ out, int n) {
  int i = blockIdx.x * blockDim.x + threadIdx.x;
  if (i < n) out[i] = (__bf16)fmaxf((float)h[i] + (float)r[i], 0.0f);
}

__global__ void k_add_relu_f32(const __bf16* __restrict__ h, const __bf16* __restrict__ r,
                               float* __restrict__ out, int n) {
  int i = blockIdx.x * blockDim.x + threadIdx.x;
  if (i < n) out[i] = fmaxf((float)h[i] + (float)r[i], 0.0f);
}

// Pack conv weight (Cout x Kd) f32 -> (Mp x Kp) bf16, zero padded.
__global__ void k_pack_w(const float* __restrict__ W, __bf16* __restrict__ Wp,
                         int M, int Kd, int Kp, int Mp) {
  int i = blockIdx.x * blockDim.x + threadIdx.x;
  int total = Mp * Kp;
  if (i >= total) return;
  int m = i / Kp, k = i % Kp;
  float v = (m < M && k < Kd) ? W[(size_t)m * Kd + k] : 0.0f;
  Wp[i] = (__bf16)v;
}

// im2col for causal dilated conv.  X: (B=1024, Cin, L=15) bf16.
// Col: (N=15360 rows) x (Kp cols, k contiguous) bf16;  k = ci*Kt + kt.
__global__ void k_im2col(const __bf16* __restrict__ X, __bf16* __restrict__ Col,
                         int Cin, int Kt, int dil, int Kd, int Kp) {
  long i = (long)blockIdx.x * blockDim.x + threadIdx.x;
  long total = 15360L * Kp;
  if (i >= total) return;
  int k = (int)(i % Kp);
  int n = (int)(i / Kp);
  int b = n / 15, l = n % 15;
  __bf16 v = (__bf16)0.0f;
  if (k < Kd) {
    int ci = k / Kt, kt = k % Kt;
    int pos = l + (kt - (Kt - 1)) * dil;   // left-causal pad of (Kt-1)*dil
    if (pos >= 0) v = X[((long)b * Cin + ci) * 15 + pos];
  }
  Col[i] = v;
}

// ---------------- conv GEMM (1 M-tile per wave; used for small Mp) ----------
__global__ void k_gemm_bf16(const __bf16* __restrict__ A, const __bf16* __restrict__ Bc,
                            const float* __restrict__ bias, __bf16* __restrict__ out,
                            int M, int Mp, int Kp, int relu) {
  const int N = 15360;
  int gid  = blockIdx.x * blockDim.x + threadIdx.x;
  int wave = gid >> 5, lane = gid & 31;
  int tilesM = Mp >> 4, tilesN = N >> 4;
  int tm = wave % tilesM, tn = wave / tilesM;
  if (tn >= tilesN) return;
  int half = lane >> 4, lm = lane & 15;
  const __bf16* ap = A  + (size_t)(tm * 16 + lm) * Kp + half * 8;
  const __bf16* bp = Bc + (size_t)(tn * 16 + lm) * Kp + half * 8;
  v8f acc = {};
  for (int k0 = 0; k0 < Kp; k0 += 32) {
    __builtin_prefetch(bp + k0 + 128, 0, 0);   // global_prefetch_b8
    v8bf alo = *(const v8bf*)(ap + k0);
    v8bf ahi = *(const v8bf*)(ap + k0 + 16);
    v8bf blo = *(const v8bf*)(bp + k0);
    v8bf bhi = *(const v8bf*)(bp + k0 + 16);
    acc = WMMA_BF16(SHUF16(alo, ahi), SHUF16(blo, bhi), acc);
  }
  int n = tn * 16 + lm, b = n / 15, l = n % 15;
#pragma unroll
  for (int r = 0; r < 8; ++r) {
    int m = tm * 16 + r + half * 8;
    if (m < M) {
      float v = acc[r] + bias[m];
      if (relu) v = fmaxf(v, 0.0f);
      out[((size_t)b * M + m) * 15 + l] = (__bf16)v;
    }
  }
}

// ---------------- conv GEMM, 4 M-tiles per wave (Mp multiple of 64) ---------
// One B (Col) fragment shared across 4 A fragments -> 4x less Col L2 traffic.
__global__ void k_gemm_bf16x4(const __bf16* __restrict__ A, const __bf16* __restrict__ Bc,
                              const float* __restrict__ bias, __bf16* __restrict__ out,
                              int M, int Mp, int Kp, int relu) {
  const int N = 15360;
  int gid  = blockIdx.x * blockDim.x + threadIdx.x;
  int wave = gid >> 5, lane = gid & 31;
  int tiles4 = Mp >> 6, tilesN = N >> 4;
  int tm = wave % tiles4, tn = wave / tiles4;
  if (tn >= tilesN) return;
  int half = lane >> 4, lm = lane & 15;
  const __bf16* bp = Bc + (size_t)(tn * 16 + lm) * Kp + half * 8;
  v8f acc[4];
  v8f z = {};
#pragma unroll
  for (int i = 0; i < 4; ++i) acc[i] = z;
  for (int k0 = 0; k0 < Kp; k0 += 32) {
    __builtin_prefetch(bp + k0 + 128, 0, 0);
    v8bf blo = *(const v8bf*)(bp + k0);
    v8bf bhi = *(const v8bf*)(bp + k0 + 16);
    v16bf bv = SHUF16(blo, bhi);
#pragma unroll
    for (int i = 0; i < 4; ++i) {
      const __bf16* ap = A + (size_t)(tm * 64 + i * 16 + lm) * Kp + half * 8 + k0;
      v8bf alo = *(const v8bf*)ap;
      v8bf ahi = *(const v8bf*)(ap + 16);
      acc[i] = WMMA_BF16(SHUF16(alo, ahi), bv, acc[i]);
    }
  }
  int n = tn * 16 + lm, b = n / 15, l = n % 15;
#pragma unroll
  for (int i = 0; i < 4; ++i) {
#pragma unroll
    for (int r = 0; r < 8; ++r) {
      int m = tm * 64 + i * 16 + r + half * 8;
      if (m < M) {
        float v = acc[i][r] + bias[m];
        if (relu) v = fmaxf(v, 0.0f);
        out[((size_t)b * M + m) * 15 + l] = (__bf16)v;
      }
    }
  }
}

// ---------------- attention (constant over scan steps) ----------------
// v[f] = sum_h w2[h] * W1[h, 2H + f]
__global__ void k_attn_v(const float* __restrict__ w1, const float* __restrict__ w2,
                         float* __restrict__ v) {
  int f = threadIdx.x;
  if (f < 15) {
    float s = 0.0f;
    for (int h = 0; h < 512; ++h) s += w2[h] * w1[(size_t)h * 1039 + 1024 + f];
    v[f] = s;
  }
}

// Per-sample: score=enc.v, softmax over t=63, context, ctfc = ctx.fc_w[:15]+fc_b.
__global__ void k_attn_ctx(const float* __restrict__ enc, const float* __restrict__ v,
                           const float* __restrict__ fcw, const float* __restrict__ fcb,
                           float* __restrict__ ctx, float* __restrict__ ctfc) {
  __shared__ float s[64];
  __shared__ float beta[64];
  __shared__ float red[2];
  __shared__ float cf[16];
  int b = blockIdx.x, t = threadIdx.x;
  const float* e = enc + (size_t)b * 63 * 15;
  float sc = -1e30f;
  if (t < 63) {
    float a = 0.0f;
#pragma unroll
    for (int f = 0; f < 15; ++f) a += e[t * 15 + f] * v[f];
    sc = a;
  }
  s[t] = sc;
  __syncthreads();
  if (t == 0) {
    float m = -1e30f;
    for (int i = 0; i < 63; ++i) m = fmaxf(m, s[i]);
    red[0] = m;
  }
  __syncthreads();
  beta[t] = (t < 63) ? __expf(s[t] - red[0]) : 0.0f;
  __syncthreads();
  if (t == 0) {
    float sm = 0.0f;
    for (int i = 0; i < 63; ++i) sm += beta[i];
    red[1] = 1.0f / sm;
  }
  __syncthreads();
  if (t < 15) {
    float c = 0.0f;
    for (int tt = 0; tt < 63; ++tt) c += beta[tt] * red[1] * e[tt * 15 + t];
    ctx[b * 15 + t] = c;
    cf[t] = c * fcw[t];
  }
  __syncthreads();
  if (t == 0) {
    float a = fcb[0];
    for (int f = 0; f < 15; ++f) a += cf[f];
    ctfc[b] = a;
  }
}

// ---------------- LSTM step: gates = d@Whh^T + y_tilde*wih + bih + bhh ----------
// 4 batch-tiles x 4 gate-tiles per wave (16 WMMA accumulators): each Whh
// fragment is reused by 4 A fragments -> Whh L2 traffic cut 4x per step.
__global__ void k_lstm_step(const __bf16* __restrict__ din, const __bf16* __restrict__ whh,
                            const float* __restrict__ wih, const float* __restrict__ bih,
                            const float* __restrict__ bhh, const float* __restrict__ ctfc,
                            const float* __restrict__ yprev, const float* __restrict__ fcw,
                            float* __restrict__ c, float* __restrict__ dF,
                            __bf16* __restrict__ dout, int t) {
  const int H = 512, B = 1024;
  int gid  = blockIdx.x * blockDim.x + threadIdx.x;
  int wave = gid >> 5, lane = gid & 31;
  int tilesB4 = B >> 6;                          // 16
  int tb = wave % tilesB4, th = wave / tilesB4;  // th in [0,32)
  if (th >= (H >> 4)) return;
  int half = lane >> 4, lm = lane & 15;
  int hcol = th * 16 + lm;
  v8f acc[4][4];                                 // [gate][batch-subtile]
  v8f z = {};
#pragma unroll
  for (int g = 0; g < 4; ++g)
#pragma unroll
    for (int i = 0; i < 4; ++i) acc[g][i] = z;

  for (int k0 = 0; k0 < H; k0 += 32) {
    v16bf bv[4];
#pragma unroll
    for (int g = 0; g < 4; ++g) {
      const __bf16* bp = whh + (size_t)(g * H + hcol) * H + half * 8 + k0;
      v8bf lo = *(const v8bf*)bp;
      v8bf hi = *(const v8bf*)(bp + 16);
      bv[g] = SHUF16(lo, hi);
    }
#pragma unroll
    for (int i = 0; i < 4; ++i) {
      const __bf16* ap = din + (size_t)(tb * 64 + i * 16 + lm) * H + half * 8 + k0;
      v8bf lo = *(const v8bf*)ap;
      v8bf hi = *(const v8bf*)(ap + 16);
      v16bf av = SHUF16(lo, hi);
#pragma unroll
      for (int g = 0; g < 4; ++g) acc[g][i] = WMMA_BF16(av, bv[g], acc[g][i]);
    }
  }

  float wy  = fcw[15];
  int   h   = hcol;
  float wi0 = wih[h],        wi1 = wih[512 + h];
  float wi2 = wih[1024 + h], wi3 = wih[1536 + h];
  float bb0 = bih[h]        + bhh[h];
  float bb1 = bih[512 + h]  + bhh[512 + h];
  float bb2 = bih[1024 + h] + bhh[1024 + h];
  float bb3 = bih[1536 + h] + bhh[1536 + h];
#pragma unroll
  for (int i = 0; i < 4; ++i) {
#pragma unroll
    for (int r = 0; r < 8; ++r) {
      int b = tb * 64 + i * 16 + r + half * 8;
      float yt = ctfc[b] + wy * yprev[b * 63 + t];
      float gi = acc[0][i][r] + yt * wi0 + bb0;
      float gf = acc[1][i][r] + yt * wi1 + bb1;
      float gg = acc[2][i][r] + yt * wi2 + bb2;
      float go = acc[3][i][r] + yt * wi3 + bb3;
      size_t idx = (size_t)b * H + h;
      float cn = sigmoidf_(gf) * c[idx] + sigmoidf_(gi) * tanhf(gg);
      c[idx] = cn;
      float dn = sigmoidf_(go) * tanhf(cn);
      dF[idx]   = dn;
      dout[idx] = (__bf16)dn;
    }
  }
}

// ---------------- final: y[b] = fcfw[:512].dT[b] + fcfw[512:527].ctx[b] + fcfb ----
__global__ void k_final(const float* __restrict__ dT, const float* __restrict__ ctx,
                        const float* __restrict__ fcfw, const float* __restrict__ fcfb,
                        float* __restrict__ out) {
  int gid  = blockIdx.x * blockDim.x + threadIdx.x;
  int wave = gid >> 5, lane = gid & 31;
  int b = wave;
  if (b >= 1024) return;
  float s = 0.0f;
  for (int h = lane; h < 512; h += 32) s += fcfw[h] * dT[(size_t)b * 512 + h];
  if (lane < 15) s += fcfw[512 + lane] * ctx[b * 15 + lane];
#pragma unroll
  for (int off = 16; off; off >>= 1) s += __shfl_xor(s, off, 32);
  if (lane == 0) out[b] = s + fcfb[0];
}

// ---------------------------------------------------------------------------
extern "C" void kernel_launch(void* const* d_in, const int* in_sizes, int n_in,
                              void* d_out, int out_size, void* d_ws, size_t ws_size,
                              hipStream_t stream) {
  (void)in_sizes; (void)n_in; (void)out_size; (void)ws_size;
  const float* X     = (const float*)d_in[0];
  const float* yprev = (const float*)d_in[1];
  const float* aw1   = (const float*)d_in[22];
  const float* aw2   = (const float*)d_in[24];
  const float* fc_w  = (const float*)d_in[26];
  const float* fc_b  = (const float*)d_in[27];
  const float* wih   = (const float*)d_in[28];
  const float* whh   = (const float*)d_in[29];
  const float* bihp  = (const float*)d_in[30];
  const float* bhhp  = (const float*)d_in[31];
  const float* fcfw  = (const float*)d_in[32];
  const float* fcfb  = (const float*)d_in[33];

  // workspace layout (~124 MB)
  char* w = (char*)d_ws;
  constexpr size_t SZ_ACT = 7864320ull * 2;        // (B,512,15) bf16
  __bf16* act0 = (__bf16*)(w);
  __bf16* act1 = (__bf16*)(w + 1 * SZ_ACT);
  __bf16* act2 = (__bf16*)(w + 2 * SZ_ACT);
  __bf16* act3 = (__bf16*)(w + 3 * SZ_ACT);
  __bf16* col  = (__bf16*)(w + 4 * SZ_ACT);        // 15360 x 1536 bf16
  size_t off = 4 * SZ_ACT + 47185920ull;
  __bf16* wpk  = (__bf16*)(w + off); off += 512ull * 1536 * 2;
  __bf16* whhb = (__bf16*)(w + off); off += 2048ull * 512 * 2;
  __bf16* dbf0 = (__bf16*)(w + off); off += 1024ull * 512 * 2;
  __bf16* dbf1 = (__bf16*)(w + off); off += 1024ull * 512 * 2;
  float*  cst  = (float*)(w + off);  off += 1024ull * 512 * 4;
  float*  dFp  = (float*)(w + off);  off += 1024ull * 512 * 4;
  float*  enc  = (float*)(w + off);  off += 967680ull * 4;
  float*  vv   = (float*)(w + off);  off += 256;
  float*  ctx  = (float*)(w + off);  off += 15360ull * 4;
  float*  ctfc = (float*)(w + off);  off += 4096;

  auto g1 = [](long n) { return dim3((unsigned)((n + 255) / 256)); };

  auto run_conv = [&](const float* Wf, const float* bias, const __bf16* Xin, __bf16* Out,
                      int Cin, int Cout, int Kt, int dil, int relu) {
    int Kd = Cin * Kt, Kp = (Kd + 31) & ~31, Mp = (Cout + 15) & ~15;
    k_pack_w<<<g1((long)Mp * Kp), 256, 0, stream>>>(Wf, wpk, Cout, Kd, Kp, Mp);
    k_im2col<<<g1(15360L * Kp), 256, 0, stream>>>(Xin, col, Cin, Kt, dil, Kd, Kp);
    if (Mp >= 256 && (Mp & 63) == 0) {
      long waves = (long)(Mp / 64) * 960;        // N/16 = 960
      k_gemm_bf16x4<<<g1(waves * 32), 256, 0, stream>>>(wpk, col, bias, Out, Cout, Mp, Kp, relu);
    } else {
      long waves = (long)(Mp / 16) * 960;
      k_gemm_bf16<<<g1(waves * 32), 256, 0, stream>>>(wpk, col, bias, Out, Cout, Mp, Kp, relu);
    }
  };

  // ---- TCN ----
  k_f32_to_bf16<<<g1(967680), 256, 0, stream>>>(X, act0, 967680);
  // block 0 (dil=1, downsample 63->512)
  run_conv((const float*)d_in[2],  (const float*)d_in[3],  act0, act1, 63,  512, 3, 1, 1);
  run_conv((const float*)d_in[4],  (const float*)d_in[5],  act1, act2, 512, 512, 3, 1, 1);
  run_conv((const float*)d_in[6],  (const float*)d_in[7],  act0, act3, 63,  512, 1, 1, 0);
  k_add_relu_bf<<<g1(7864320), 256, 0, stream>>>(act2, act3, act0, 7864320);
  // block 1 (dil=2, identity residual)
  run_conv((const float*)d_in[8],  (const float*)d_in[9],  act0, act1, 512, 512, 3, 2, 1);
  run_conv((const float*)d_in[10], (const float*)d_in[11], act1, act2, 512, 512, 3, 2, 1);
  k_add_relu_bf<<<g1(7864320), 256, 0, stream>>>(act2, act0, act0, 7864320);
  // block 2 (dil=4, identity residual)
  run_conv((const float*)d_in[12], (const float*)d_in[13], act0, act1, 512, 512, 3, 4, 1);
  run_conv((const float*)d_in[14], (const float*)d_in[15], act1, act2, 512, 512, 3, 4, 1);
  k_add_relu_bf<<<g1(7864320), 256, 0, stream>>>(act2, act0, act0, 7864320);
  // block 3 (dil=8, downsample 512->63); result -> encod f32
  run_conv((const float*)d_in[16], (const float*)d_in[17], act0, act1, 512, 63, 3, 8, 1);
  run_conv((const float*)d_in[18], (const float*)d_in[19], act1, act2, 63,  63, 3, 8, 1);
  run_conv((const float*)d_in[20], (const float*)d_in[21], act0, act3, 512, 63, 1, 1, 0);
  k_add_relu_f32<<<g1(967680), 256, 0, stream>>>(act2, act3, enc, 967680);

  // ---- attention collapses to a single precompute (softmax shift-invariance) ----
  k_attn_v<<<1, 32, 0, stream>>>(aw1, aw2, vv);
  k_attn_ctx<<<1024, 64, 0, stream>>>(enc, vv, fc_w, fc_b, ctx, ctfc);

  // ---- LSTM scan: 63 dependent steps ----
  k_f32_to_bf16<<<g1(1048576), 256, 0, stream>>>(whh, whhb, 1048576);
  k_zero_u32<<<g1(524288), 256, 0, stream>>>((unsigned int*)cst, 524288);
  k_zero_u32<<<g1(262144), 256, 0, stream>>>((unsigned int*)dbf0, 262144);
  for (int t = 0; t < 63; ++t) {
    const __bf16* din = (t & 1) ? dbf1 : dbf0;
    __bf16*       dou = (t & 1) ? dbf0 : dbf1;
    k_lstm_step<<<64, 256, 0, stream>>>(din, whhb, wih, bihp, bhhp, ctfc, yprev, fc_w,
                                        cst, dFp, dou, t);
  }

  // ---- y_pred = [dT, context] @ fcf_w^T + fcf_b ----
  k_final<<<128, 256, 0, stream>>>(dFp, ctx, fcfw, fcfb, (float*)d_out);
}
